// CrossMultiHeadAttention_7404523618319
// MI455X (gfx1250) — matrix operand order repaired and checked by reading, not compiled, not run on hardware
//
#include <hip/hip_runtime.h>
#include <hip/hip_bf16.h>

#define L_DIM 1024
#define N_DIM 8
#define E_DIM 1024
#define H_DIM 8

typedef __attribute__((ext_vector_type(16))) __bf16 v16bf;
typedef __attribute__((ext_vector_type(8)))  float  v8f;

union FragAB { v16bf v; uint4 u[2]; };

// output modes for the shared WMMA GEMM engine
#define OUT_QK   0   // bf16, [h][n][l][d]   (rows r = l*N+n)
#define OUT_V    1   // bf16, [h][n][d][s]   (rows r = s*N+n, transposed store)
#define OUT_F32  2   // fp32, row-major with ldc (scores / final out)
#define OUT_CTX  3   // bf16, [h][l][n][d]   (per-(h,n) GEMM, rows = l)
#define OUT_HEAD 4   // bf16, concat[r][h*E + c]

// Async copy: 16B from global directly into LDS (ASYNCcnt-tracked).
__device__ __forceinline__ void async_copy_b128(unsigned lds_off, const void* gptr)
{
    asm volatile("global_load_async_to_lds_b128 %0, %1, off"
                 :: "v"(lds_off), "v"(gptr)
                 : "memory");
}
__device__ __forceinline__ void wait_async0()
{
    asm volatile("s_wait_asynccnt 0x0" ::: "memory");
}

// ---------------------------------------------------------------------------
// Generic C = A * B^T (+bias) GEMM. A: (M x K) rows, B: (N x K) rows, both
// bf16 with K contiguous. 256 threads = 8 waves; 128x256 block tile; each
// wave computes 64x64 via 4x4 v_wmma_f32_16x16x32_bf16 tiles; K-step 32.
// Tiles stream global->LDS via double-buffered global_load_async_to_lds_b128
// (one barrier per K-step, copy overlapped with WMMA).
// LDS rows padded to 40 bf16 (= 5 uint4) for bank spread.
// ---------------------------------------------------------------------------
template <int MODE>
__global__ __launch_bounds__(256)
void gemm_wmma(const __hip_bfloat16* __restrict__ A,
               const __hip_bfloat16* __restrict__ B,
               const float* __restrict__ bias,
               float* __restrict__ outF,
               __hip_bfloat16* __restrict__ outB,
               int M, int Nn, int Kk,
               long strideAz, long strideBz, long strideBiasZ,
               long strideOz, int ldc, float scale)
{
    __shared__ uint4 lA4[2][128 * 5];   // 128 rows x 32 bf16 (+8 pad), x2 buffers
    __shared__ uint4 lB4[2][256 * 5];   // 256 rows x 32 bf16 (+8 pad), x2 buffers

    const int t    = threadIdx.x;
    const int lane = t & 31;
    const int w    = t >> 5;      // wave 0..7
    const int wr   = w >> 2;      // 0..1  (wave row: 64 M-rows each)
    const int wc   = w & 3;       // 0..3  (wave col: 64 N-cols each)
    const int z    = blockIdx.z;

    const __hip_bfloat16* Az = A + (size_t)z * strideAz + (size_t)blockIdx.y * 128 * Kk;
    const __hip_bfloat16* Bz = B + (size_t)z * strideBz + (size_t)blockIdx.x * 256 * Kk;
    const float* biasz = bias ? (bias + (size_t)z * strideBiasZ + (size_t)blockIdx.x * 256)
                              : nullptr;

    v8f acc[4][4];
#pragma unroll
    for (int mi = 0; mi < 4; ++mi)
#pragma unroll
        for (int ni = 0; ni < 4; ++ni)
            acc[mi][ni] = (v8f){0.f, 0.f, 0.f, 0.f, 0.f, 0.f, 0.f, 0.f};

    // per-thread staging coordinates (16B chunks, 4 chunks per 32-col row)
    const int cs = t & 3;
    const unsigned baseA = (unsigned)(uintptr_t)&lA4[0][0];
    const unsigned baseB = (unsigned)(uintptr_t)&lB4[0][0];

    auto stage = [&](int buf, int k0) {
        const unsigned boA = (unsigned)buf * (128u * 5u * 16u);
        const unsigned boB = (unsigned)buf * (256u * 5u * 16u);
        // A tile: 128 rows -> 512 chunks -> 2 per thread
#pragma unroll
        for (int j = 0; j < 2; ++j) {
            const int r = (t + j * 256) >> 2;
            async_copy_b128(baseA + boA + (unsigned)(r * 5 + cs) * 16u,
                            Az + (size_t)r * Kk + k0 + cs * 8);
        }
        // B tile: 256 rows -> 1024 chunks -> 4 per thread
#pragma unroll
        for (int j = 0; j < 4; ++j) {
            const int r = (t + j * 256) >> 2;
            async_copy_b128(baseB + boB + (unsigned)(r * 5 + cs) * 16u,
                            Bz + (size_t)r * Kk + k0 + cs * 8);
        }
    };

    const int kq  = lane >> 4;    // 0/1 : which K-half this lane holds
    const int r16 = lane & 15;
    const int kSteps = Kk >> 5;

    stage(0, 0);                               // prologue: tile 0 in flight

    for (int ks = 0; ks < kSteps; ++ks) {
        const int buf = ks & 1;

        wait_async0();        // own async copies (into buf) done
        __syncthreads();      // everyone's copies done => buf complete;
                              // also: all reads of buf^1 finished last step

        if (ks + 1 < kSteps)  // stream next tile while we compute
            stage(buf ^ 1, (ks + 1) << 5);

        // ---- load fragments per ISA 7.12.2 layouts
        FragAB fa[4], fb[4];
#pragma unroll
        for (int mi = 0; mi < 4; ++mi) {
            const uint4* p = &lA4[buf][(wr * 64 + mi * 16 + r16) * 5 + kq];
            fa[mi].u[0] = p[0];   // K = koff .. koff+7
            fa[mi].u[1] = p[2];   // K = koff+16 .. koff+23
        }
#pragma unroll
        for (int ni = 0; ni < 4; ++ni) {
            const uint4* p = &lB4[buf][(wc * 64 + ni * 16 + r16) * 5 + kq];
            fb[ni].u[0] = p[0];   // K = kq*8 .. kq*8+7
            fb[ni].u[1] = p[2];   // K = 16+kq*8 .. 16+kq*8+7
        }

#pragma unroll
        for (int mi = 0; mi < 4; ++mi)
#pragma unroll
            for (int ni = 0; ni < 4; ++ni)
                acc[mi][ni] = __builtin_amdgcn_wmma_f32_16x16x32_bf16(
                    false, fa[mi].v, false, fb[ni].v,
                    (short)0, acc[mi][ni], false, false);
    }

    // ---- store: C 16x16 layout: VGPR i -> (M = i + 8*(lane>=16), N = lane&15)
    const int hi = lane >> 4;
    const int lc = lane & 15;
#pragma unroll
    for (int mi = 0; mi < 4; ++mi) {
#pragma unroll
        for (int ni = 0; ni < 4; ++ni) {
            const int m0 = blockIdx.y * 128 + wr * 64 + mi * 16;
            const int n0 = blockIdx.x * 256 + wc * 64 + ni * 16;
            const int gc = n0 + lc;
            const float bv = biasz ? biasz[wc * 64 + ni * 16 + lc] : 0.f;
#pragma unroll
            for (int i = 0; i < 8; ++i) {
                const int gr = m0 + hi * 8 + i;
                const float val = acc[mi][ni][i] * scale + bv;
                if (MODE == OUT_F32) {
                    outF[(size_t)z * strideOz + (size_t)gr * ldc + gc] = val;
                } else if (MODE == OUT_QK) {
                    const int l = gr >> 3, n = gr & 7;
                    outB[(((size_t)z * N_DIM + n) * L_DIM + l) * E_DIM + gc] =
                        __float2bfloat16(val);
                } else if (MODE == OUT_V) {
                    const int s = gr >> 3, n = gr & 7;
                    outB[(((size_t)z * N_DIM + n) * E_DIM + gc) * L_DIM + s] =
                        __float2bfloat16(val);
                } else if (MODE == OUT_CTX) {
                    const int h = z >> 3, n = z & 7;
                    outB[(((size_t)h * L_DIM + gr) * N_DIM + n) * E_DIM + gc] =
                        __float2bfloat16(val);
                } else { // OUT_HEAD
                    outB[(size_t)gr * (H_DIM * E_DIM) + (size_t)z * E_DIM + gc] =
                        __float2bfloat16(val);
                }
            }
        }
    }
}

// ---------------------------------------------------------------------------
// fp32 -> bf16 bulk convert (n multiple of 4)
// ---------------------------------------------------------------------------
__global__ __launch_bounds__(256)
void cvt_kernel(const float* __restrict__ in, __hip_bfloat16* __restrict__ out, long n)
{
    long i = ((long)blockIdx.x * 256 + threadIdx.x) * 4;
    if (i < n) {
        float4 f = *(const float4*)(in + i);
        out[i + 0] = __float2bfloat16(f.x);
        out[i + 1] = __float2bfloat16(f.y);
        out[i + 2] = __float2bfloat16(f.z);
        out[i + 3] = __float2bfloat16(f.w);
    }
}

// ---------------------------------------------------------------------------
// In-place row softmax over 1024 fp32 scores + bf16 copy for the ctx GEMM.
// One 256-thread block per row (4 elems/thread).
// ---------------------------------------------------------------------------
__global__ __launch_bounds__(256)
void softmax_kernel(float* __restrict__ attn, __hip_bfloat16* __restrict__ P)
{
    __shared__ float red[256];
    const int t = threadIdx.x;
    float* x = attn + (size_t)blockIdx.x * L_DIM;

    float v[4];
    float m = -1e30f;
#pragma unroll
    for (int k = 0; k < 4; ++k) { v[k] = x[t + k * 256]; m = fmaxf(m, v[k]); }
    red[t] = m;
    __syncthreads();
    for (int s = 128; s > 0; s >>= 1) {
        if (t < s) red[t] = fmaxf(red[t], red[t + s]);
        __syncthreads();
    }
    m = red[0];
    __syncthreads();

    float sum = 0.f;
#pragma unroll
    for (int k = 0; k < 4; ++k) { v[k] = __expf(v[k] - m); sum += v[k]; }
    red[t] = sum;
    __syncthreads();
    for (int s = 128; s > 0; s >>= 1) {
        if (t < s) red[t] += red[t + s];
        __syncthreads();
    }
    const float inv = 1.0f / red[0];

    __hip_bfloat16* p = P + (size_t)blockIdx.x * L_DIM;
#pragma unroll
    for (int k = 0; k < 4; ++k) {
        const float r = v[k] * inv;
        x[t + k * 256] = r;
        p[t + k * 256] = __float2bfloat16(r);
    }
}

// ---------------------------------------------------------------------------
extern "C" void kernel_launch(void* const* d_in, const int* in_sizes, int n_in,
                              void* d_out, int out_size, void* d_ws, size_t ws_size,
                              hipStream_t stream)
{
    (void)in_sizes; (void)n_in; (void)out_size; (void)ws_size;

    const float* query = (const float*)d_in[0];
    const float* key   = (const float*)d_in[1];
    const float* value = (const float*)d_in[2];
    const float* Wqkv  = (const float*)d_in[3];
    const float* bqkv  = (const float*)d_in[4];
    const float* Wo    = (const float*)d_in[5];
    const float* bo    = (const float*)d_in[6];
    const float* Wout  = (const float*)d_in[7];
    const float* bout  = (const float*)d_in[8];

    const long LN   = (long)L_DIM * N_DIM;                  // 8192
    const long LNE  = LN * E_DIM;                           // 8.4M
    const long HNLE = (long)H_DIM * N_DIM * L_DIM * E_DIM;  // 64M
    const long HLL  = (long)H_DIM * N_DIM * L_DIM * L_DIM;  // 64M (attn)

    float* outp  = (float*)d_out;
    float* attnp = outp + LNE;          // attn region of d_out (fp32, hnls)

    // workspace carve-out (bf16 elements)
    __hip_bfloat16* ws = (__hip_bfloat16*)d_ws;
    size_t o = 0;
    auto carve = [&](long e) { __hip_bfloat16* p = ws + o; o += (size_t)e; return p; };
    __hip_bfloat16* XQ   = carve(LNE);
    __hip_bfloat16* XK   = carve(LNE);
    __hip_bfloat16* XV   = carve(LNE);
    __hip_bfloat16* WQKV = carve((long)H_DIM * 3 * E_DIM * E_DIM);
    __hip_bfloat16* WOb  = carve((long)H_DIM * E_DIM * E_DIM);
    __hip_bfloat16* WOUT = carve((long)E_DIM * H_DIM * E_DIM);
    __hip_bfloat16* Q    = carve(HNLE);
    __hip_bfloat16* Kt   = carve(HNLE);
    __hip_bfloat16* Vt   = carve(HNLE);
    __hip_bfloat16* P    = carve(HLL);
    __hip_bfloat16* CTX    = Q;   // Q dead after scores GEMM
    __hip_bfloat16* CONCAT = Kt;  // K dead after scores GEMM

    dim3 blk(256);

    // 0) fp32 -> bf16 conversions
    cvt_kernel<<<dim3((unsigned)(LNE / 1024)), blk, 0, stream>>>(query, XQ, LNE);
    cvt_kernel<<<dim3((unsigned)(LNE / 1024)), blk, 0, stream>>>(key,   XK, LNE);
    cvt_kernel<<<dim3((unsigned)(LNE / 1024)), blk, 0, stream>>>(value, XV, LNE);
    cvt_kernel<<<dim3((unsigned)(((long)H_DIM * 3 * E_DIM * E_DIM) / 1024)), blk, 0, stream>>>(
        Wqkv, WQKV, (long)H_DIM * 3 * E_DIM * E_DIM);
    cvt_kernel<<<dim3((unsigned)(((long)H_DIM * E_DIM * E_DIM) / 1024)), blk, 0, stream>>>(
        Wo, WOb, (long)H_DIM * E_DIM * E_DIM);
    cvt_kernel<<<dim3((unsigned)(((long)E_DIM * H_DIM * E_DIM) / 1024)), blk, 0, stream>>>(
        Wout, WOUT, (long)E_DIM * H_DIM * E_DIM);

    const long EE3 = 3L * E_DIM * E_DIM;

    // 1) QKV projections: per head, (LN x E) = X(LN x E) * Wsec(E x E)^T + b
    gemm_wmma<OUT_QK><<<dim3(E_DIM / 256, LN / 128, H_DIM), blk, 0, stream>>>(
        XQ, WQKV + 0L * E_DIM * E_DIM, bqkv + 0L * E_DIM, nullptr, Q,
        (int)LN, E_DIM, E_DIM, 0, EE3, 3L * E_DIM, 0, 0, 1.0f);
    gemm_wmma<OUT_QK><<<dim3(E_DIM / 256, LN / 128, H_DIM), blk, 0, stream>>>(
        XK, WQKV + 1L * E_DIM * E_DIM, bqkv + 1L * E_DIM, nullptr, Kt,
        (int)LN, E_DIM, E_DIM, 0, EE3, 3L * E_DIM, 0, 0, 1.0f);
    gemm_wmma<OUT_V><<<dim3(E_DIM / 256, LN / 128, H_DIM), blk, 0, stream>>>(
        XV, WQKV + 2L * E_DIM * E_DIM, bqkv + 2L * E_DIM, nullptr, Vt,
        (int)LN, E_DIM, E_DIM, 0, EE3, 3L * E_DIM, 0, 0, 1.0f);

    // 2) scores = scale * Q K^T per (h,n), straight into d_out attn region
    gemm_wmma<OUT_F32><<<dim3(L_DIM / 256, L_DIM / 128, H_DIM * N_DIM), blk, 0, stream>>>(
        Q, Kt, nullptr, attnp, nullptr,
        L_DIM, L_DIM, E_DIM,
        (long)L_DIM * E_DIM, (long)L_DIM * E_DIM, 0,
        (long)L_DIM * L_DIM, L_DIM, 0.03125f /* 1/sqrt(1024) */);

    // 3) softmax in place (+ bf16 P)
    softmax_kernel<<<dim3(H_DIM * N_DIM * L_DIM), blk, 0, stream>>>(attnp, P);

    // 4) ctx = P * V  (B = V^T rows, so A*B^T form) -> ctx[h][l][n][d]
    gemm_wmma<OUT_CTX><<<dim3(E_DIM / 256, L_DIM / 128, H_DIM * N_DIM), blk, 0, stream>>>(
        P, Vt, nullptr, nullptr, CTX,
        L_DIM, E_DIM, L_DIM,
        (long)L_DIM * L_DIM, (long)E_DIM * L_DIM, 0, 0, 0, 1.0f);

    // 5) head_out = ctx * Wo^T + bo -> concat[r][h*E + e]
    gemm_wmma<OUT_HEAD><<<dim3(E_DIM / 256, LN / 128, H_DIM), blk, 0, stream>>>(
        CTX, WOb, bo, nullptr, CONCAT,
        (int)LN, E_DIM, E_DIM,
        (long)LN * E_DIM, (long)E_DIM * E_DIM, E_DIM, 0, 0, 1.0f);

    // 6) out = concat * Wout^T + bout  (fp32 to d_out)
    gemm_wmma<OUT_F32><<<dim3(E_DIM / 256, LN / 128, 1), blk, 0, stream>>>(
        CONCAT, WOUT, bout, outp, nullptr,
        (int)LN, E_DIM, H_DIM * E_DIM,
        0, 0, 0, 0, E_DIM, 1.0f);
}
